// SCCA_38139309588693
// MI455X (gfx1250) — matrix operand order repaired and checked by reading, not compiled, not run on hardware
//
#include <hip/hip_runtime.h>
#include <hip/hip_bf16.h>

// ---------------------------------------------------------------------------
// SCCA (criss-cross attention + spatial/channel gating) for MI455X (gfx1250)
// Heavy matmuls run on v_wmma_f32_16x16x32_f16 (wave32). LDS tiles are stored
// pre-swizzled into the CDNA5 per-lane fragment layout: every WMMA operand is
// fetched with 2x ds_load_b128; staging uses 16B global_load_b128 and, where
// the swizzle allows, 16B ds_store_b128.
// ---------------------------------------------------------------------------

typedef __attribute__((ext_vector_type(16))) _Float16 v16h;
typedef __attribute__((ext_vector_type(8)))  _Float16 v8h;
typedef __attribute__((ext_vector_type(8)))  float    v8f;

// ---- CDNA5 16x16x32 f16 fragment packing (ISA 7.12.2, wave32) -------------
// A tile 16x32 (m,k) -> frag[lane][slot], lane = m + ((k>>3 & 1)<<4),
//                                         slot = (k&7) + ((k>>4)<<3)
__device__ __forceinline__ int a_slot_idx(int m, int k) {
  int lane = m + (((k >> 3) & 1) << 4);
  int slot = (k & 7) + ((k >> 4) << 3);
  return lane * 16 + slot;
}
// B tile 32x16 (k,n) -> frag[lane][slot], lane = n + ((k>>3 & 1)<<4),
//                                         slot = (k&7) + ((k>>4)<<3)
__device__ __forceinline__ int b_slot_idx(int k, int n) {
  int lane = n + (((k >> 3) & 1) << 4);
  int slot = (k & 7) + ((k >> 4) << 3);
  return lane * 16 + slot;
}

__device__ __forceinline__ v16h frag_ld(const _Float16* fragbase, int lane) {
  return *(const v16h*)(fragbase + lane * 16);  // 32B contiguous per lane
}

// ---------------------------------------------------------------------------
// Elementwise / reduction helper kernels
// ---------------------------------------------------------------------------

__global__ __launch_bounds__(256)
void k_f32_to_f16(const float* __restrict__ in, _Float16* __restrict__ out, size_t n) {
  for (size_t i = (size_t)blockIdx.x * blockDim.x + threadIdx.x; i < n;
       i += (size_t)gridDim.x * blockDim.x)
    out[i] = (_Float16)in[i];
}

// logits[b,n] = dot(wvec[b*wstride + :], x[b,:,n])   (wstride==0 => shared w)
__global__ __launch_bounds__(256)
void k_pixdot(const float* __restrict__ X, const float* __restrict__ Wv,
              float* __restrict__ L, int Npix, int wstride) {
  size_t idx = (size_t)blockIdx.x * blockDim.x + threadIdx.x;
  size_t total = (size_t)16 * Npix;
  if (idx >= total) return;
  int b = (int)(idx / Npix), n = (int)(idx % Npix);
  const float* xb = X + (size_t)b * 256 * Npix + n;
  const float* wv = Wv + (size_t)b * wstride;
  float s = 0.f;
#pragma unroll 8
  for (int c = 0; c < 256; ++c) s += wv[c] * xb[(size_t)c * Npix];
  L[idx] = s;
}

// per-batch softmax over Npix; mode 0: softmax, mode 1: sigmoid(softmax)
__global__ __launch_bounds__(256)
void k_softmax_gate(const float* __restrict__ L, float* __restrict__ Outp,
                    int Npix, int mode) {
  __shared__ float red[256];
  __shared__ float stat;
  int b = blockIdx.x;
  const float* Lb = L + (size_t)b * Npix;
  float mx = -3.0e38f;
  for (int n = threadIdx.x; n < Npix; n += 256) mx = fmaxf(mx, Lb[n]);
  red[threadIdx.x] = mx;
  __syncthreads();
  for (int s = 128; s; s >>= 1) {
    if ((int)threadIdx.x < s) red[threadIdx.x] = fmaxf(red[threadIdx.x], red[threadIdx.x + s]);
    __syncthreads();
  }
  if (!threadIdx.x) stat = red[0];
  __syncthreads();
  float m = stat, sum = 0.f;
  for (int n = threadIdx.x; n < Npix; n += 256) sum += __expf(Lb[n] - m);
  __syncthreads();
  red[threadIdx.x] = sum;
  __syncthreads();
  for (int s = 128; s; s >>= 1) {
    if ((int)threadIdx.x < s) red[threadIdx.x] += red[threadIdx.x + s];
    __syncthreads();
  }
  if (!threadIdx.x) stat = red[0];
  __syncthreads();
  float inv = 1.f / stat;
  for (int n = threadIdx.x; n < Npix; n += 256) {
    float v = __expf(Lb[n] - m) * inv;
    Outp[(size_t)b * Npix + n] = mode ? 1.f / (1.f + __expf(-v)) : v;
  }
}

// Sv[b,c] = sum_n X[b,c,n] * Mw[b,n]   (Mw==nullptr => mean over n)
__global__ __launch_bounds__(256)
void k_chanreduce(const float* __restrict__ X, const float* __restrict__ Mw,
                  float* __restrict__ Sv, int Npix) {
  __shared__ float red[256];
  int bc = blockIdx.x;
  int b  = bc >> 8;
  const float* xr = X + (size_t)bc * Npix;
  const float* mr = Mw ? (Mw + (size_t)b * Npix) : nullptr;
  float invn = 1.0f / (float)Npix;
  float s = 0.f;
  for (int n = threadIdx.x; n < Npix; n += 256)
    s += xr[n] * (mr ? mr[n] : invn);
  red[threadIdx.x] = s;
  __syncthreads();
  for (int st = 128; st; st >>= 1) {
    if ((int)threadIdx.x < st) red[threadIdx.x] += red[threadIdx.x + st];
    __syncthreads();
  }
  if (!threadIdx.x) Sv[bc] = red[0];
}

// spatial gate: ctx = Wvr @ s ; g = sigmoid(Wup @ ctx + bup)
__global__ __launch_bounds__(256)
void k_gate_spatial(const float* __restrict__ Sv, const float* __restrict__ Wvr,
                    const float* __restrict__ Wup, const float* __restrict__ Bup,
                    float* __restrict__ G) {
  __shared__ float sh_s[256], sh_ctx[128];
  int b = blockIdx.x, tid = threadIdx.x;
  sh_s[tid] = Sv[b * 256 + tid];
  __syncthreads();
  if (tid < 128) {
    float a = 0.f;
    for (int c = 0; c < 256; ++c) a += Wvr[tid * 256 + c] * sh_s[c];
    sh_ctx[tid] = a;
  }
  __syncthreads();
  float a = Bup[tid];
  for (int j = 0; j < 128; ++j) a += Wup[tid * 128 + j] * sh_ctx[j];
  G[b * 256 + tid] = 1.f / (1.f + __expf(-a));
}

// channel gate vector: avg = Wql @ xbar ; u = Wvl^T @ avg
__global__ __launch_bounds__(256)
void k_gate_channel(const float* __restrict__ Xbar, const float* __restrict__ Wql,
                    const float* __restrict__ Wvl, float* __restrict__ U) {
  __shared__ float sh_x[256], sh_avg[128];
  int b = blockIdx.x, tid = threadIdx.x;
  sh_x[tid] = Xbar[b * 256 + tid];
  __syncthreads();
  if (tid < 128) {
    float a = 0.f;
    for (int c = 0; c < 256; ++c) a += Wql[tid * 256 + c] * sh_x[c];
    sh_avg[tid] = a;
  }
  __syncthreads();
  float a = 0.f;
  for (int j = 0; j < 128; ++j) a += Wvl[j * 256 + tid] * sh_avg[j];
  U[b * 256 + tid] = a;
}

__global__ __launch_bounds__(256)
void k_scale_chan(const float* __restrict__ X, const float* __restrict__ G,
                  float* __restrict__ T, size_t total, int Npix) {
  for (size_t i = (size_t)blockIdx.x * blockDim.x + threadIdx.x; i < total;
       i += (size_t)gridDim.x * blockDim.x) {
    int c = (int)((i / Npix) & 255);
    int b = (int)(i / ((size_t)Npix * 256));
    T[i] = X[i] * G[b * 256 + c];
  }
}

__global__ __launch_bounds__(256)
void k_scale_pix(const float* __restrict__ X, const float* __restrict__ G,
                 float* __restrict__ T, size_t total, int Npix) {
  for (size_t i = (size_t)blockIdx.x * blockDim.x + threadIdx.x; i < total;
       i += (size_t)gridDim.x * blockDim.x) {
    int n = (int)(i % Npix);
    int b = (int)(i / ((size_t)Npix * 256));
    T[i] = X[i] * G[(size_t)b * Npix + n];
  }
}

__global__ __launch_bounds__(256)
void k_add(const float* __restrict__ A, const float* __restrict__ B,
           float* __restrict__ O, size_t n) {
  for (size_t i = (size_t)blockIdx.x * blockDim.x + threadIdx.x; i < n;
       i += (size_t)gridDim.x * blockDim.x)
    O[i] = A[i] + B[i];
}

// y = gamma * (oh + ow) + t
__global__ __launch_bounds__(256)
void k_combine(const float* __restrict__ OH, const float* __restrict__ OW,
               const float* __restrict__ T, const float* __restrict__ gamma,
               float* __restrict__ Y, size_t n) {
  float g = gamma[0];
  for (size_t i = (size_t)blockIdx.x * blockDim.x + threadIdx.x; i < n;
       i += (size_t)gridDim.x * blockDim.x)
    Y[i] = g * (OH[i] + OW[i]) + T[i];
}

// ---------------------------------------------------------------------------
// WMMA kernels (fragment-swizzled LDS staging, 16B vectorized)
// ---------------------------------------------------------------------------

// conv1x1 GEMM: Y16[b,o,n] = f16( sum_c W[o,c]*T[b,c,n] + bias[o] )
// grid: (Npix/64, O/16, B), block 128 (4 waves, one 16x16 n-subtile each)
__global__ __launch_bounds__(128)
void k_conv1x1_wmma(const _Float16* __restrict__ Wt, const _Float16* __restrict__ T,
                    const float* __restrict__ bias, _Float16* __restrict__ Y,
                    int O, int Npix) {
  __shared__ __align__(32) _Float16 As[512];       // one A tile, frag-packed
  __shared__ __align__(32) _Float16 Bs[4 * 512];   // 4 B tiles, frag-packed
  int lane = threadIdx.x & 31, wave = threadIdx.x >> 5;
  int n0 = blockIdx.x * 64;
  int m0 = blockIdx.y * 16;
  int b  = blockIdx.z;
  const _Float16* Tb = T + (size_t)b * 256 * Npix;
  v8f acc = {};
  for (int kc = 0; kc < 256; kc += 32) {
    // A tile (weights): 64 chunks of 8 halves; vector load + vector LDS store
    if (threadIdx.x < 64) {
      int m = threadIdx.x >> 2, seg = threadIdx.x & 3;
      v8h av = *(const v8h*)(Wt + (m0 + m) * 256 + kc + seg * 8);
      int k0 = seg * 8;
      int lane_d = m + (((k0 >> 3) & 1) << 4);
      int slot0  = (k0 >> 4) << 3;
      *(v8h*)(As + lane_d * 16 + slot0) = av;
    }
    // B tile (activations): 256 chunks of 8 halves; vector load + scalar scatter
#pragma unroll
    for (int j = 0; j < 2; ++j) {
      int chunk = threadIdx.x + 128 * j;
      int k = chunk >> 3, seg = chunk & 7;
      v8h bv = *(const v8h*)(Tb + (size_t)(kc + k) * Npix + n0 + seg * 8);
#pragma unroll
      for (int e = 0; e < 8; ++e) {
        int n = seg * 8 + e;
        Bs[(n >> 4) * 512 + b_slot_idx(k, n & 15)] = bv[e];
      }
    }
    __syncthreads();
    v16h a  = frag_ld(As, lane);
    v16h bf = frag_ld(Bs + wave * 512, lane);
    acc = __builtin_amdgcn_wmma_f32_16x16x32_f16(false, a, false, bf, (short)0,
                                                 acc, false, false);
    __syncthreads();
  }
  int n = lane & 15, mb = (lane < 16) ? 0 : 8;
#pragma unroll
  for (int r = 0; r < 8; ++r) {
    int m = m0 + mb + r;
    float vb = bias ? bias[m] : 0.f;
    Y[((size_t)b * O + m) * Npix + n0 + wave * 16 + n] = (_Float16)(acc[r] + vb);
  }
}

// Criss-cross score GEMM. dir 0: fixed h=line (row / e_w); dir 1: fixed w=line
// (col / e_h). Computes S = Q^T K (96x96, K=32) with 36 WMMAs per block.
__global__ __launch_bounds__(128)
void k_scores_wmma(const _Float16* __restrict__ Q, const _Float16* __restrict__ K,
                   float* __restrict__ E, int dir) {
  __shared__ __align__(32) _Float16 Qs[6 * 512];   // 6 A tiles (m-tiles)
  __shared__ __align__(32) _Float16 Ks[6 * 512];   // 6 B tiles (n-tiles)
  int line = blockIdx.x % 96, b = blockIdx.x / 96;
  int lane = threadIdx.x & 31, wave = threadIdx.x >> 5;
  const size_t HW = 96 * 96;
  const _Float16* Qb = Q + (size_t)b * 32 * HW;
  const _Float16* Kb = K + (size_t)b * 32 * HW;
  if (dir == 0) {
    // rows contiguous in p: 384 chunks of 8 halves, 3 per thread
#pragma unroll
    for (int j = 0; j < 3; ++j) {
      int chunk = threadIdx.x + 128 * j;
      int kk = chunk / 12, seg = chunk % 12;
      size_t off = (size_t)kk * HW + (size_t)line * 96 + seg * 8;
      v8h qv = *(const v8h*)(Qb + off);
      v8h kv = *(const v8h*)(Kb + off);
#pragma unroll
      for (int e = 0; e < 8; ++e) {
        int p = seg * 8 + e;
        Qs[(p >> 4) * 512 + a_slot_idx(p & 15, kk)] = qv[e];
        Ks[(p >> 4) * 512 + b_slot_idx(kk, p & 15)] = kv[e];
      }
    }
  } else {
    for (int i = threadIdx.x; i < 32 * 96; i += 128) {
      int kk = i / 96, p = i % 96;
      size_t off = (size_t)kk * HW + (size_t)p * 96 + line;
      Qs[(p >> 4) * 512 + a_slot_idx(p & 15, kk)] = Qb[off];
      Ks[(p >> 4) * 512 + b_slot_idx(kk, p & 15)] = Kb[off];
    }
  }
  __syncthreads();
  for (int t = wave; t < 36; t += 4) {
    int mi = t / 6, ni = t % 6;
    v16h a  = frag_ld(Qs + mi * 512, lane);
    v16h bf = frag_ld(Ks + ni * 512, lane);
    v8f acc = {};
    acc = __builtin_amdgcn_wmma_f32_16x16x32_f16(false, a, false, bf, (short)0,
                                                 acc, false, false);
    int n = ni * 16 + (lane & 15), mb = (lane < 16) ? 0 : 8;
#pragma unroll
    for (int r = 0; r < 8; ++r) {
      int m = mi * 16 + mb + r;
      size_t o = (dir == 0)
                     ? (((size_t)(b * 96 + line) * 96 + m) * 96 + n)  // EW[b][h][w][v]
                     : (((size_t)(b * 96 + m) * 96 + line) * 96 + n); // EH[b][h][w][k]
      E[o] = acc[r];
    }
  }
}

// softmax over concat(e_h (diag-masked), e_w), length 192, one wave per pixel
__global__ __launch_bounds__(128)
void k_attn_softmax(const float* __restrict__ EH, const float* __restrict__ EW,
                    _Float16* __restrict__ AH, _Float16* __restrict__ AW, int npix) {
  int lane = threadIdx.x & 31, wave = threadIdx.x >> 5;
  int p = blockIdx.x * 4 + wave;
  if (p >= npix) return;
  int h = (p / 96) % 96;
  size_t base = (size_t)p * 96;
  float vh[3], vw[3], mx = -3.0e38f;
#pragma unroll
  for (int j = 0; j < 3; ++j) {
    int idx = lane + 32 * j;
    float a = EH[base + idx];
    if (idx == h) a = -3.0e38f;  // diagonal mask on the H branch
    vh[j] = a;
    float w = EW[base + idx];
    vw[j] = w;
    mx = fmaxf(mx, fmaxf(a, w));
  }
#pragma unroll
  for (int s = 16; s >= 1; s >>= 1) mx = fmaxf(mx, __shfl_xor(mx, s, 32));
  float sum = 0.f;
#pragma unroll
  for (int j = 0; j < 3; ++j) {
    vh[j] = __expf(vh[j] - mx);
    vw[j] = __expf(vw[j] - mx);
    sum += vh[j] + vw[j];
  }
#pragma unroll
  for (int s = 16; s >= 1; s >>= 1) sum += __shfl_xor(sum, s, 32);
  float inv = 1.f / sum;
#pragma unroll
  for (int j = 0; j < 3; ++j) {
    AH[base + lane + 32 * j] = (_Float16)(vh[j] * inv);
    AW[base + lane + 32 * j] = (_Float16)(vw[j] * inv);
  }
}

// attention-apply GEMM: dir 0 (out_w, fixed h=line): O[c][w] = sum_v V[c,h,v]*Aw[h,w,v]
//                       dir 1 (out_h, fixed w=line): O[c][h] = sum_k V[c,k,w]*Ah[h,w,k]
// grid: (B*96, 4 channel groups of 64), block 256 (8 waves, 3 tiles each)
__global__ __launch_bounds__(256)
void k_out_wmma(const _Float16* __restrict__ V, const _Float16* __restrict__ Att,
                float* __restrict__ Out, int dir) {
  __shared__ __align__(32) _Float16 Vs[12 * 512];   // (mi,kc) A tiles: 4x3
  __shared__ __align__(32) _Float16 As_[18 * 512];  // (ni,kc) B tiles: 6x3
  int line = blockIdx.x % 96, b = blockIdx.x / 96;
  int c0 = blockIdx.y * 64;
  int lane = threadIdx.x & 31, wave = threadIdx.x >> 5;
  const size_t HW = 96 * 96;
  const _Float16* Vb = V + (size_t)b * 256 * HW;
  if (dir == 0) {
    // V rows contiguous: 768 chunks of 8 halves; vector load + vector LDS store
#pragma unroll
    for (int j = 0; j < 3; ++j) {
      int chunk = threadIdx.x + 256 * j;
      int mc = chunk / 12, seg = chunk % 12;
      v8h vv = *(const v8h*)(Vb + (size_t)(c0 + mc) * HW + (size_t)line * 96 + seg * 8);
      int kk0 = seg * 8;
      int kc = kk0 >> 5, kloc0 = kk0 & 31;
      int lane_d = (mc & 15) + (((kloc0 >> 3) & 1) << 4);
      int slot0  = (kloc0 >> 4) << 3;
      *(v8h*)(Vs + ((mc >> 4) * 3 + kc) * 512 + lane_d * 16 + slot0) = vv;
    }
  } else {
    for (int i = threadIdx.x; i < 64 * 96; i += 256) {
      int mc = i / 96, kk = i % 96;
      _Float16 v = Vb[(size_t)(c0 + mc) * HW + (size_t)kk * 96 + line];
      Vs[((mc >> 4) * 3 + (kk >> 5)) * 512 + a_slot_idx(mc & 15, kk & 31)] = v;
    }
  }
  // attention rows contiguous in kk for both dirs: 1152 chunks of 8 halves;
  // vector load + vector LDS store (B layout: lane const, slots consecutive)
  for (int chunk = threadIdx.x; chunk < 1152; chunk += 256) {
    int np = chunk / 12, seg = chunk % 12;
    size_t o = (dir == 0) ? (((size_t)(b * 96 + line) * 96 + np) * 96 + seg * 8)
                          : (((size_t)(b * 96 + np) * 96 + line) * 96 + seg * 8);
    v8h av = *(const v8h*)(Att + o);
    int kk0 = seg * 8;
    int kc = kk0 >> 5, kloc0 = kk0 & 31;
    int lane_d = (np & 15) + (((kloc0 >> 3) & 1) << 4);
    int slot0  = (kloc0 >> 4) << 3;  // 0 or 8
    *(v8h*)(As_ + ((np >> 4) * 3 + kc) * 512 + lane_d * 16 + slot0) = av;
  }
  __syncthreads();
  v8f acc[3] = {};
  for (int kc = 0; kc < 3; ++kc) {
#pragma unroll
    for (int t = 0; t < 3; ++t) {
      int tid = wave * 3 + t;
      int mi = tid / 6, ni = tid % 6;
      v16h a  = frag_ld(Vs + (mi * 3 + kc) * 512, lane);
      v16h bf = frag_ld(As_ + (ni * 3 + kc) * 512, lane);
      acc[t] = __builtin_amdgcn_wmma_f32_16x16x32_f16(false, a, false, bf, (short)0,
                                                      acc[t], false, false);
    }
  }
  int n_ = lane & 15, mb = (lane < 16) ? 0 : 8;
#pragma unroll
  for (int t = 0; t < 3; ++t) {
    int tid = wave * 3 + t, mi = tid / 6, ni = tid % 6;
#pragma unroll
    for (int r = 0; r < 8; ++r) {
      int c = c0 + mi * 16 + mb + r;
      int pos = ni * 16 + n_;
      size_t o = (dir == 0)
                     ? (((size_t)(b * 256 + c) * 96 + line) * 96 + pos)
                     : (((size_t)(b * 256 + c) * 96 + pos) * 96 + line);
      Out[o] = acc[t][r];
    }
  }
}

// ---------------------------------------------------------------------------
// Host orchestration
// ---------------------------------------------------------------------------

extern "C" void kernel_launch(void* const* d_in, const int* in_sizes, int n_in,
                              void* d_out, int out_size, void* d_ws, size_t ws_size,
                              hipStream_t stream) {
  (void)in_sizes; (void)n_in; (void)out_size; (void)ws_size;
  const int B_ = 16, C_ = 256, CQ_ = 32, N_ = 96 * 96;
  const size_t ECN  = (size_t)B_ * C_ * N_;
  const size_t EQN  = (size_t)B_ * CQ_ * N_;
  const size_t EATT = (size_t)B_ * N_ * 96;

  const float* x        = (const float*)d_in[0];
  const float* w_q_right= (const float*)d_in[1];
  const float* w_v_right= (const float*)d_in[2];
  const float* w_up     = (const float*)d_in[3];
  const float* b_up     = (const float*)d_in[4];
  const float* w_q_left = (const float*)d_in[5];
  const float* w_v_left = (const float*)d_in[6];
  const float* cr_wq    = (const float*)d_in[7];
  const float* cr_bq    = (const float*)d_in[8];
  const float* cr_wk    = (const float*)d_in[9];
  const float* cr_bk    = (const float*)d_in[10];
  const float* cr_wv    = (const float*)d_in[11];
  const float* cr_bv    = (const float*)d_in[12];
  const float* gamma    = (const float*)d_in[13];
  float* out            = (float*)d_out;

  char* p = (char*)d_ws;
  auto carve = [&](size_t bytes) -> void* {
    void* r = (void*)p;
    p += (bytes + 255) & ~(size_t)255;
    return r;
  };
  float*    bufA  = (float*)carve(ECN * 4);
  float*    bufB  = (float*)carve(ECN * 4);
  float*    bufC  = (float*)carve(ECN * 4);
  _Float16* T16   = (_Float16*)carve(ECN * 2);
  _Float16* Q16   = (_Float16*)carve(EQN * 2);
  _Float16* K16   = (_Float16*)carve(EQN * 2);
  _Float16* V16   = (_Float16*)carve(ECN * 2);
  float*    EH    = (float*)carve(EATT * 4);
  float*    EW    = (float*)carve(EATT * 4);
  _Float16* AH16  = (_Float16*)carve(EATT * 2);
  _Float16* AW16  = (_Float16*)carve(EATT * 2);
  float*    OH    = (float*)carve(ECN * 4);
  float*    OW    = (float*)carve(ECN * 4);
  _Float16* WQ16  = (_Float16*)carve((size_t)CQ_ * C_ * 2);
  _Float16* WK16  = (_Float16*)carve((size_t)CQ_ * C_ * 2);
  _Float16* WV16  = (_Float16*)carve((size_t)C_ * C_ * 2);
  float*    logits= (float*)carve((size_t)B_ * N_ * 4);
  float*    mbuf  = (float*)carve((size_t)B_ * N_ * 4);
  float*    gpix  = (float*)carve((size_t)B_ * N_ * 4);
  float*    svec  = (float*)carve((size_t)B_ * C_ * 4);
  float*    xbar  = (float*)carve((size_t)B_ * C_ * 4);
  float*    gatec = (float*)carve((size_t)B_ * C_ * 4);
  float*    uvec  = (float*)carve((size_t)B_ * C_ * 4);

  // one full criss-cross attention pass: yout = gamma*(out_h+out_w) + tin
  auto cross = [&](const float* tin, float* yout) {
    k_f32_to_f16<<<2048, 256, 0, stream>>>(tin, T16, ECN);
    dim3 gq(N_ / 64, CQ_ / 16, B_);
    k_conv1x1_wmma<<<gq, 128, 0, stream>>>(WQ16, T16, cr_bq, Q16, CQ_, N_);
    k_conv1x1_wmma<<<gq, 128, 0, stream>>>(WK16, T16, cr_bk, K16, CQ_, N_);
    dim3 gv(N_ / 64, C_ / 16, B_);
    k_conv1x1_wmma<<<gv, 128, 0, stream>>>(WV16, T16, cr_bv, V16, C_, N_);
    k_scores_wmma<<<B_ * 96, 128, 0, stream>>>(Q16, K16, EW, 0);
    k_scores_wmma<<<B_ * 96, 128, 0, stream>>>(Q16, K16, EH, 1);
    k_attn_softmax<<<(B_ * N_) / 4, 128, 0, stream>>>(EH, EW, AH16, AW16, B_ * N_);
    dim3 go(B_ * 96, 4);
    k_out_wmma<<<go, 256, 0, stream>>>(V16, AW16, OW, 0);
    k_out_wmma<<<go, 256, 0, stream>>>(V16, AH16, OH, 1);
    k_combine<<<2048, 256, 0, stream>>>(OH, OW, tin, gamma, yout, ECN);
  };

  // cross-attention weights -> f16 (once per call; deterministic)
  k_f32_to_f16<<<32, 256, 0, stream>>>(cr_wq, WQ16, (size_t)CQ_ * C_);
  k_f32_to_f16<<<32, 256, 0, stream>>>(cr_wk, WK16, (size_t)CQ_ * C_);
  k_f32_to_f16<<<256, 256, 0, stream>>>(cr_wv, WV16, (size_t)C_ * C_);

  // ---- spatial_weighted(x) -> bufA ----
  k_pixdot<<<(B_ * N_ + 255) / 256, 256, 0, stream>>>(x, w_q_right, logits, N_, 0);
  k_softmax_gate<<<B_, 256, 0, stream>>>(logits, mbuf, N_, 0);
  k_chanreduce<<<B_ * C_, 256, 0, stream>>>(x, mbuf, svec, N_);
  k_gate_spatial<<<B_, 256, 0, stream>>>(svec, w_v_right, w_up, b_up, gatec);
  k_scale_chan<<<2048, 256, 0, stream>>>(x, gatec, bufA, ECN, N_);

  // ---- ctx_channel = cross(spatial_weighted(x)) -> bufB ----
  cross(bufA, bufB);

  // ---- channel_weighted(x) -> bufA ----
  k_chanreduce<<<B_ * C_, 256, 0, stream>>>(x, nullptr, xbar, N_);
  k_gate_channel<<<B_, 256, 0, stream>>>(xbar, w_q_left, w_v_left, uvec);
  k_pixdot<<<(B_ * N_ + 255) / 256, 256, 0, stream>>>(x, uvec, logits, N_, C_);
  k_softmax_gate<<<B_, 256, 0, stream>>>(logits, gpix, N_, 1);
  k_scale_pix<<<2048, 256, 0, stream>>>(x, gpix, bufA, ECN, N_);

  // ---- ctx_spatial = cross(channel_weighted(x)) -> bufC ----
  cross(bufA, bufC);

  // ---- final: cross(ctx_spatial + ctx_channel) -> out ----
  k_add<<<2048, 256, 0, stream>>>(bufB, bufC, bufA, ECN);
  cross(bufA, out);
}
